// ContinuousNeuralField_15152644620828
// MI455X (gfx1250) — matrix operand order repaired and checked by reading, not compiled, not run on hardware
//
#include <hip/hip_runtime.h>

typedef __attribute__((ext_vector_type(16))) __bf16 v16bf;
typedef __attribute__((ext_vector_type(8)))  __bf16 v8bf;
typedef __attribute__((ext_vector_type(8)))  float  v8f;

#define NN    8000
#define BB    64
#define IND   784
#define INDP  800   // 784 zero-padded to a multiple of 32 (K for bf16 WMMA)
#define OUTD  10
#define FDIM  32
#define VOLF  100.0f

static constexpr size_t algn(size_t x){ return (x + 255) & ~size_t(255); }
static constexpr size_t OFF_CW    = 0;                                       // bf16 [NN][NN]
static constexpr size_t OFF_RS    = algn(OFF_CW    + (size_t)NN*NN*2);       // f32  [NN]
static constexpr size_t OFF_ACTA  = algn(OFF_RS    + (size_t)NN*4);          // f32  [NN][BB]
static constexpr size_t OFF_ACTB  = algn(OFF_ACTA  + (size_t)NN*BB*4);       // f32  [NN][BB]
static constexpr size_t OFF_ACTTA = algn(OFF_ACTB  + (size_t)NN*BB*4);       // bf16 [BB][NN]
static constexpr size_t OFF_ACTTB = algn(OFF_ACTTA + (size_t)BB*NN*2);       // bf16 [BB][NN]
static constexpr size_t OFF_FNB   = algn(OFF_ACTTB + (size_t)BB*NN*2);       // bf16 [NN][FDIM]
static constexpr size_t OFF_POSC  = algn(OFF_FNB   + (size_t)NN*FDIM*2);     // f32  [NN][4]
static constexpr size_t OFF_RADC  = algn(OFF_POSC  + (size_t)NN*4*4);        // f32  [NN]
static constexpr size_t OFF_IWU   = algn(OFF_RADC  + (size_t)NN*4);          // f32  [NN]
static constexpr size_t OFF_OWU   = algn(OFF_IWU   + (size_t)NN*4);          // f32  [NN]
static constexpr size_t OFF_WB    = algn(OFF_OWU   + (size_t)NN*4);          // bf16 [NN][INDP]
static constexpr size_t OFF_XB    = algn(OFF_WB    + (size_t)NN*INDP*2);     // bf16 [BB][INDP]
static constexpr size_t OFF_SCAL  = algn(OFF_XB    + (size_t)BB*INDP*2);     // f32  [4]

__device__ __forceinline__ v8f wmma_bf16(v16bf a, v16bf b, v8f c) {
  return __builtin_amdgcn_wmma_f32_16x16x32_bf16(false, a, false, b, (short)0, c,
                                                 false, false);
}
__device__ __forceinline__ v16bf pack16(v8bf lo, v8bf hi) {
  v16bf r;
  #pragma unroll
  for (int q = 0; q < 8; ++q) { r[q] = lo[q]; r[8+q] = hi[q]; }
  return r;
}

// ---------------------------------------------------------------- init
__global__ void k_init(float* __restrict__ rowsum, float* __restrict__ scal) {
  int g = blockIdx.x * blockDim.x + threadIdx.x;
  if (g < NN) rowsum[g] = 0.0f;
  if (g < 4)  scal[g]   = 0.0f;
}

// ---------------------------------------------------------------- prep
__global__ void k_prep(const float* __restrict__ positions,
                       const float* __restrict__ radii,
                       const float* __restrict__ features,
                       float* __restrict__ posc, float* __restrict__ radc,
                       float* __restrict__ iwu,  float* __restrict__ owu,
                       __bf16* __restrict__ fnb, float* __restrict__ scal) {
  int n = blockIdx.x * blockDim.x + threadIdx.x;
  if (n >= NN) return;
  float px = fminf(fmaxf(positions[n*3+0], 0.1f), VOLF - 0.1f);
  float py = fminf(fmaxf(positions[n*3+1], 0.1f), VOLF - 0.1f);
  float pz = fminf(fmaxf(positions[n*3+2], 0.1f), VOLF - 0.1f);
  posc[n*4+0] = px; posc[n*4+1] = py; posc[n*4+2] = pz;
  posc[n*4+3] = px*px + py*py + pz*pz;
  float rc = fminf(fmaxf(radii[n], 1.0f), 50.0f);
  radc[n] = rc;
  atomicMax((int*)&scal[2], __float_as_int(rc));   // positive floats: int order == float order
  float xc = fminf(fmaxf(px * (1.0f/VOLF), 0.0f), 1.0f);
  float iw = __expf(-3.0f * xc);
  float ow = __expf( 3.0f * (xc - 1.0f));
  iwu[n] = iw; owu[n] = ow;
  atomicAdd(&scal[0], iw);
  atomicAdd(&scal[1], ow);
  float s = 0.0f;
  for (int k = 0; k < FDIM; ++k) { float f = features[n*FDIM+k]; s += f*f; }
  float inv = 1.0f / fmaxf(sqrtf(s), 1e-6f);
  for (int k = 0; k < FDIM; ++k)
    fnb[n*FDIM+k] = (__bf16)(features[n*FDIM+k] * inv);
}

// ---------------------------------------------------------------- fp32 -> padded bf16 copies
__global__ void k_convW(const float* __restrict__ W, __bf16* __restrict__ Wb) {
  long g = (long)blockIdx.x * blockDim.x + threadIdx.x;
  if (g >= (long)NN * INDP) return;
  int n = (int)(g / INDP), k = (int)(g % INDP);
  Wb[g] = (k < IND) ? (__bf16)W[(long)n*IND + k] : (__bf16)0.0f;
}
__global__ void k_convX(const float* __restrict__ x, __bf16* __restrict__ xb) {
  int g = blockIdx.x * blockDim.x + threadIdx.x;
  if (g >= BB * INDP) return;
  int b = g / INDP, k = g % INDP;
  xb[g] = (k < IND) ? (__bf16)x[b*IND + k] : (__bf16)0.0f;
}

// ---------------------------------------------------------------- input projection GEMM
// act0[n][b] = (sum_k W[n,k]*x[b,k]) * iw[n]/(sum_iw+1e-6)
// WMMA: M=n, N=b, K=800. No LDS: A rows (Wb) and B columns (xb rows) are both
// K-contiguous, so fragments come straight from global_load_b128 pairs.
__global__ void __launch_bounds__(128)
k_ingemm(const __bf16* __restrict__ Wb, const __bf16* __restrict__ xb,
         const float* __restrict__ iwu, const float* __restrict__ scal,
         float* __restrict__ actaF, __bf16* __restrict__ actaT) {
  int tid = threadIdx.x, wv = tid >> 5, lane = tid & 31;
  int i0 = blockIdx.x * 32;
  int mi = wv >> 1, nbp = wv & 1;               // wave: one A frag, two b-tiles
  int nb0 = nbp * 2, nb1 = nb0 + 1;
  v8f c0 = {}, c1 = {};
  int aoff = (lane < 16) ? 0 : 8;
  int kro  = (lane < 16) ? 0 : 8;
  const __bf16* arp = Wb + (size_t)(i0 + mi*16 + (lane & 15)) * INDP + aoff;
  const __bf16* bp0 = xb + (size_t)(nb0*16 + (lane & 15)) * INDP + kro;
  const __bf16* bp1 = xb + (size_t)(nb1*16 + (lane & 15)) * INDP + kro;
  for (int kc = 0; kc < INDP; kc += 32) {
    v16bf a = pack16(*(const v8bf*)(arp + kc),      *(const v8bf*)(arp + kc + 16));
    v16bf b0 = pack16(*(const v8bf*)(bp0 + kc),     *(const v8bf*)(bp0 + kc + 16));
    v16bf b1 = pack16(*(const v8bf*)(bp1 + kc),     *(const v8bf*)(bp1 + kc + 16));
    c0 = wmma_bf16(a, b0, c0);
    c1 = wmma_bf16(a, b1, c1);
  }
  float inv_siw = 1.0f / (scal[0] + 1e-6f);
  #pragma unroll
  for (int r = 0; r < 8; ++r) {
    int m  = r + ((lane < 16) ? 0 : 8);
    int gn = i0 + mi*16 + m;
    float s = iwu[gn] * inv_siw;
    int b0c = nb0*16 + (lane & 15), b1c = nb1*16 + (lane & 15);
    float v0 = c0[r] * s, v1 = c1[r] * s;
    actaF[(size_t)gn*BB + b0c] = v0;
    actaF[(size_t)gn*BB + b1c] = v1;
    actaT[(size_t)b0c*NN + gn] = (__bf16)v0;
    actaT[(size_t)b1c*NN + gn] = (__bf16)v1;
  }
}

// ---------------------------------------------------------------- connection matrix build
// cw[i][j] (bf16, unnormalized) + rowsum[i]. feat_sim via WMMA (K=FDIM=32);
// feature fragments read directly from L2-resident fnb, pos/radii via LDS.
__global__ void __launch_bounds__(256)
k_cw(const __bf16* __restrict__ fnb, const float* __restrict__ posc,
     const float* __restrict__ radc, const float* __restrict__ scal,
     __bf16* __restrict__ cw, float* __restrict__ rowsum) {
  __shared__ float  pi_[64][4];
  __shared__ float  pj_[64][4];
  __shared__ float  ri_[64];
  int tid = threadIdx.x, wv = tid >> 5, lane = tid & 31;
  int i0 = blockIdx.y * 64, j0 = blockIdx.x * 64;
  for (int e = tid; e < 256; e += 256) {
    pi_[e>>2][e&3] = posc[(size_t)(i0 + (e>>2))*4 + (e&3)];
    pj_[e>>2][e&3] = posc[(size_t)(j0 + (e>>2))*4 + (e&3)];
  }
  if (tid < 64) ri_[tid] = radc[i0 + tid];
  __syncthreads();

  int mi = wv >> 1, nbase = (wv & 1) * 2;       // wave: one A frag, two j-tiles
  int aoff = (lane < 16) ? 0 : 8;
  int kro  = (lane < 16) ? 0 : 8;
  const __bf16* arp = fnb + (size_t)(i0 + mi*16 + (lane & 15)) * FDIM + aoff;
  int lj0 = nbase*16 + (lane & 15), lj1 = lj0 + 16;
  const __bf16* bp0 = fnb + (size_t)(j0 + lj0) * FDIM + kro;
  const __bf16* bp1 = fnb + (size_t)(j0 + lj1) * FDIM + kro;
  v16bf a  = pack16(*(const v8bf*)(arp), *(const v8bf*)(arp + 16));
  v16bf b0 = pack16(*(const v8bf*)(bp0), *(const v8bf*)(bp0 + 16));
  v16bf b1 = pack16(*(const v8bf*)(bp1), *(const v8bf*)(bp1 + 16));
  v8f c0 = {}, c1 = {};
  c0 = wmma_bf16(a, b0, c0);
  c1 = wmma_bf16(a, b1, c1);

  float maxr = scal[2];
  #pragma unroll
  for (int t = 0; t < 2; ++t) {
    v8f cc = t ? c1 : c0;
    int ni = nbase + t;
    #pragma unroll
    for (int r = 0; r < 8; ++r) {
      int m   = r + ((lane < 16) ? 0 : 8);
      int liL = mi*16 + m;
      int ljL = ni*16 + (lane & 15);
      int gi = i0 + liL, gj = j0 + ljL;
      float sim = fminf(fmaxf(cc[r], -1.0f), 1.0f);
      float dx = pi_[liL][0] - pj_[ljL][0];
      float dy = pi_[liL][1] - pj_[ljL][1];
      float dz = pi_[liL][2] - pj_[ljL][2];
      float d2 = dx*dx + dy*dy + dz*dz;
      float dist = (d2 > 0.0f) ? sqrtf(d2) : 0.0f;
      float conn = (dist < maxr) ? 1.0f : 0.0f;
      float at   = __expf(-fminf(dist / (ri_[liL] + 1e-6f), 20.0f)) * conn;
      float cwv  = at * (0.3f + 0.7f * sim);
      cw[(size_t)gi * NN + gj] = (__bf16)cwv;
      float s = cwv;                            // reduce 16 lanes of this row-half
      s += __shfl_xor(s, 1, 32);
      s += __shfl_xor(s, 2, 32);
      s += __shfl_xor(s, 4, 32);
      s += __shfl_xor(s, 8, 32);
      if ((lane & 15) == 0) atomicAdd(&rowsum[gi], s);
    }
  }
}

// ---------------------------------------------------------------- recurrent iteration
// anext[i][b] = min(relu(aprev[i][b] + (sum_j cw[i,j]*aprev[j][b])/(rowsum[i]+1e-6) - thr[i]), 100)
// No LDS, no barriers: A = cw rows (K-contiguous), B = actT rows (K-contiguous).
// cw (128 MB bf16) is L2-resident, so the loop is issue-limited: ~5 instr/wmma.
__global__ void __launch_bounds__(128)
k_iter(const __bf16* __restrict__ cw, const float* __restrict__ rowsum,
       const float* __restrict__ thr, const int* __restrict__ niter, int idx,
       const float* __restrict__ aprevF, const __bf16* __restrict__ aprevT,
       float* __restrict__ anextF, __bf16* __restrict__ anextT) {
  int tid = threadIdx.x;
  int i0 = blockIdx.x * 32;
  if (idx >= *niter) {                          // passthrough keeps ping-pong consistent
    for (int e = tid; e < 32*BB; e += 128)
      anextF[(size_t)i0*BB + e] = aprevF[(size_t)i0*BB + e];
    for (int e = tid; e < BB*32; e += 128) {
      int b = e >> 5, k = e & 31;
      anextT[(size_t)b*NN + i0 + k] = aprevT[(size_t)b*NN + i0 + k];
    }
    return;
  }
  int wv = tid >> 5, lane = tid & 31;
  int mi = wv >> 1, nbp = wv & 1;
  int nb0 = nbp * 2, nb1 = nb0 + 1;
  v8f c0 = {}, c1 = {};
  int aoff = (lane < 16) ? 0 : 8;
  int kro  = (lane < 16) ? 0 : 8;
  const __bf16* arp = cw + (size_t)(i0 + mi*16 + (lane & 15)) * NN + aoff;
  const __bf16* bp0 = aprevT + (size_t)(nb0*16 + (lane & 15)) * NN + kro;
  const __bf16* bp1 = aprevT + (size_t)(nb1*16 + (lane & 15)) * NN + kro;
  #pragma unroll 2
  for (int kc = 0; kc < NN; kc += 32) {
    __builtin_prefetch(arp + kc + 128, 0, 1);   // global_prefetch for upcoming cw chunk
    v16bf a  = pack16(*(const v8bf*)(arp + kc), *(const v8bf*)(arp + kc + 16));
    v16bf b0 = pack16(*(const v8bf*)(bp0 + kc), *(const v8bf*)(bp0 + kc + 16));
    v16bf b1 = pack16(*(const v8bf*)(bp1 + kc), *(const v8bf*)(bp1 + kc + 16));
    c0 = wmma_bf16(a, b0, c0);
    c1 = wmma_bf16(a, b1, c1);
  }
  #pragma unroll
  for (int r = 0; r < 8; ++r) {
    int m  = r + ((lane < 16) ? 0 : 8);
    int gi = i0 + mi*16 + m;
    float rs = 1.0f / (rowsum[gi] + 1e-6f);
    float th = thr[gi];
    int b0c = nb0*16 + (lane & 15), b1c = nb1*16 + (lane & 15);
    float v0 = aprevF[(size_t)gi*BB + b0c] + c0[r] * rs - th;
    float v1 = aprevF[(size_t)gi*BB + b1c] + c1[r] * rs - th;
    v0 = fminf(fmaxf(v0, 0.0f), 100.0f);
    v1 = fminf(fmaxf(v1, 0.0f), 100.0f);
    anextF[(size_t)gi*BB + b0c] = v0;
    anextF[(size_t)gi*BB + b1c] = v1;
    anextT[(size_t)b0c*NN + gi] = (__bf16)v0;
    anextT[(size_t)b1c*NN + gi] = (__bf16)v1;
  }
}

// ---------------------------------------------------------------- output projection
__global__ void __launch_bounds__(256)
k_out(const float* __restrict__ actaF, const float* __restrict__ owu,
      const float* __restrict__ scal, const float* __restrict__ Wout,
      float* __restrict__ out) {
  __shared__ float red[256];
  int b = blockIdx.x, tid = threadIdx.x;
  float inv = 1.0f / (scal[1] + 1e-6f);
  float acc[OUTD];
  #pragma unroll
  for (int o = 0; o < OUTD; ++o) acc[o] = 0.0f;
  for (int i = tid; i < NN; i += 256) {
    float s = actaF[(size_t)i*BB + b] * owu[i] * inv;
    #pragma unroll
    for (int o = 0; o < OUTD; ++o) acc[o] += s * Wout[(size_t)i*OUTD + o];
  }
  for (int o = 0; o < OUTD; ++o) {
    red[tid] = acc[o]; __syncthreads();
    for (int st = 128; st > 0; st >>= 1) {
      if (tid < st) red[tid] += red[tid + st];
      __syncthreads();
    }
    if (tid == 0) out[b*OUTD + o] = red[0];
    __syncthreads();
  }
}

// ---------------------------------------------------------------- launcher
extern "C" void kernel_launch(void* const* d_in, const int* in_sizes, int n_in,
                              void* d_out, int out_size, void* d_ws, size_t ws_size,
                              hipStream_t stream) {
  const float* x         = (const float*)d_in[0];
  const float* positions = (const float*)d_in[1];
  const float* Win       = (const float*)d_in[2];
  const float* features  = (const float*)d_in[3];
  const float* Wout      = (const float*)d_in[4];
  const float* radii     = (const float*)d_in[5];
  const float* thr       = (const float*)d_in[6];
  const int*   niter     = (const int*)d_in[7];

  char* w = (char*)d_ws;
  __bf16* cwm    = (__bf16*)(w + OFF_CW);
  float*  rowsum = (float*)(w + OFF_RS);
  float*  actaF  = (float*)(w + OFF_ACTA);
  float*  actbF  = (float*)(w + OFF_ACTB);
  __bf16* actaT  = (__bf16*)(w + OFF_ACTTA);
  __bf16* actbT  = (__bf16*)(w + OFF_ACTTB);
  __bf16* fnb    = (__bf16*)(w + OFF_FNB);
  float*  posc   = (float*)(w + OFF_POSC);
  float*  radc   = (float*)(w + OFF_RADC);
  float*  iwu    = (float*)(w + OFF_IWU);
  float*  owu    = (float*)(w + OFF_OWU);
  __bf16* Wb     = (__bf16*)(w + OFF_WB);
  __bf16* xb     = (__bf16*)(w + OFF_XB);
  float*  scal   = (float*)(w + OFF_SCAL);
  float*  out    = (float*)d_out;

  k_init<<<(NN + 255) / 256, 256, 0, stream>>>(rowsum, scal);
  k_prep<<<(NN + 255) / 256, 256, 0, stream>>>(positions, radii, features,
                                               posc, radc, iwu, owu, fnb, scal);
  k_convW<<<(NN * INDP + 255) / 256, 256, 0, stream>>>(Win, Wb);
  k_convX<<<(BB * INDP + 255) / 256, 256, 0, stream>>>(x, xb);
  k_ingemm<<<NN / 32, 128, 0, stream>>>(Wb, xb, iwu, scal, actaF, actaT);
  k_cw<<<dim3(NN / 64, NN / 64), 256, 0, stream>>>(fnb, posc, radc, scal, cwm, rowsum);
  // up to 4 guarded iterations (reference uses 2); passthrough keeps result in A buffers.
  k_iter<<<NN / 32, 128, 0, stream>>>(cwm, rowsum, thr, niter, 0, actaF, actaT, actbF, actbT);
  k_iter<<<NN / 32, 128, 0, stream>>>(cwm, rowsum, thr, niter, 1, actbF, actbT, actaF, actaT);
  k_iter<<<NN / 32, 128, 0, stream>>>(cwm, rowsum, thr, niter, 2, actaF, actaT, actbF, actbT);
  k_iter<<<NN / 32, 128, 0, stream>>>(cwm, rowsum, thr, niter, 3, actbF, actbT, actaF, actaT);
  k_out<<<BB, 256, 0, stream>>>(actaF, owu, scal, Wout, out);
}
